// RecurrentNeuralNetworkModel_56684978372912
// MI455X (gfx1250) — compile-verified, matrix operand order checked by reading, not run on hardware
//
#include <hip/hip_runtime.h>
#include <hip/hip_bf16.h>

#define RNN_B  64
#define RNN_T  512
#define RNN_I  256
#define RNN_H  512
#define RNN_O  256
#define RNN_BT (RNN_B * RNN_T)

typedef __attribute__((ext_vector_type(16))) _Float16 v16h;
typedef __attribute__((ext_vector_type(8)))  _Float16 v8h;
typedef __attribute__((ext_vector_type(8)))  float    v8f;

union F16Frag { v16h v; v8h h[2]; };

// CDNA5 16-bit A/B fragment layout (ISA 7.12.2): for lane = (half, lm),
// element e holds K = (e<8 ? 0 : 16) + 8*half + (e&7).
// => two contiguous 8xf16 (16B) loads at p and p+16, where
//    p = row_base + kBase + 8*half.
__device__ __forceinline__ v16h load_frag(const _Float16* __restrict__ p) {
  F16Frag f;
  f.h[0] = *(const v8h*)(p);
  f.h[1] = *(const v8h*)(p + 16);
  return f.v;
}

__device__ __forceinline__ v8f wmma16(v16h a, v16h b, v8f c) {
  return __builtin_amdgcn_wmma_f32_16x16x32_f16(
      false, a, false, b, (short)0, c, false, false);
}

// Generic C[M,N] = A[M,K] x Bt[N,K]^T + bias1 (+ bias2); f16 in, f32 out.
// Block = 256 threads = 8 waves. Block tile 32(M) x 256(N).
// Wave tile 16(M) x 64(N): one A-fragment feeds 4 WMMAs.
// Software-pipelined one k-chunk ahead so loads overlap WMMAs (partial waits).
__global__ void __launch_bounds__(256)
gemm_wmma_f16(const _Float16* __restrict__ A,    // [M,K] row-major
              const _Float16* __restrict__ Bt,   // [N,K] row-major
              const float*    __restrict__ bias1,
              const float*    __restrict__ bias2,
              float*          __restrict__ C,    // [M,N] row-major
              int M, int N, int K)
{
  const int lane = threadIdx.x & 31;
  const int wave = threadIdx.x >> 5;
  const int half = lane >> 4;
  const int lm   = lane & 15;
  const int m0 = blockIdx.y * 32  + (wave >> 2) * 16;
  const int n0 = blockIdx.x * 256 + (wave & 3) * 64;

  const _Float16* aRow = A  + (size_t)(m0 + lm) * K + 8 * half;
  const _Float16* bRow = Bt + (size_t)(n0 + lm) * K + 8 * half;

  v8f acc0 = {}, acc1 = {}, acc2 = {}, acc3 = {};

  // prologue: fragments for k = 0
  v16h a  = load_frag(aRow);
  v16h b0 = load_frag(bRow);
  v16h b1 = load_frag(bRow + (size_t)16 * K);
  v16h b2 = load_frag(bRow + (size_t)32 * K);
  v16h b3 = load_frag(bRow + (size_t)48 * K);

  for (int k = 32; k < K; k += 32) {
    // issue next chunk's loads before consuming current fragments
    v16h an = load_frag(aRow + k);
    v16h c0 = load_frag(bRow + k);
    v16h c1 = load_frag(bRow + (size_t)16 * K + k);
    v16h c2 = load_frag(bRow + (size_t)32 * K + k);
    v16h c3 = load_frag(bRow + (size_t)48 * K + k);
    __builtin_prefetch(aRow + k + 32, 0, 1);   // gfx1250 global_prefetch_b8

    acc0 = wmma16(a, b0, acc0);
    acc1 = wmma16(a, b1, acc1);
    acc2 = wmma16(a, b2, acc2);
    acc3 = wmma16(a, b3, acc3);

    a = an; b0 = c0; b1 = c1; b2 = c2; b3 = c3;
  }
  // epilogue chunk
  acc0 = wmma16(a, b0, acc0);
  acc1 = wmma16(a, b1, acc1);
  acc2 = wmma16(a, b2, acc2);
  acc3 = wmma16(a, b3, acc3);

  v8f accs[4] = {acc0, acc1, acc2, acc3};
#pragma unroll
  for (int i = 0; i < 4; ++i) {
    const int n = n0 + 16 * i + lm;
    float bs = bias1 ? bias1[n] : 0.0f;
    if (bias2) bs += bias2[n];
#pragma unroll
    for (int r = 0; r < 8; ++r) {
      C[(size_t)(m0 + r + 8 * half) * N + n] = accs[i][r] + bs;
    }
  }
}

// One recurrence step: hnew = tanh(xp[:,t,:] + hprev @ Whh^T).
// M=64, N=512, K=512. Grid = 32 blocks x 4 waves; one 16x16 tile per wave.
// All 16 A-frag loads + 16 B-frag loads are issued as one burst (single L2
// round-trip of memory-level parallelism), then 16 WMMAs run back-to-back on
// two interleaved accumulators. This is the sequential critical path, so we
// trade occupancy (~290 VGPRs, 1 wave/SIMD) for minimal per-step latency.
__global__ void __launch_bounds__(128)
rnn_step_wmma(const _Float16* __restrict__ hprev, // [B,H] f16
              const _Float16* __restrict__ whh,   // [H,H] f16, row n = output unit
              const float*    __restrict__ xp,    // [B,T,H] f32 (proj + biases)
              int t,
              _Float16*       __restrict__ hnew,  // [B,H] f16
              _Float16*       __restrict__ outH)  // [B,T,H] f16 sequence output
{
  const int lane = threadIdx.x & 31;
  const int wave = threadIdx.x >> 5;     // 0..3
  const int half = lane >> 4;
  const int lm   = lane & 15;
  const int m0 = wave * 16;              // 4 waves cover B=64
  const int n0 = blockIdx.x * 16;        // 32 blocks cover H=512

  const _Float16* aRow = hprev + (size_t)(m0 + lm) * RNN_H + 8 * half;
  const _Float16* bRow = whh   + (size_t)(n0 + lm) * RNN_H + 8 * half;

  v16h a[16], b[16];
#pragma unroll
  for (int kk = 0; kk < 16; ++kk) a[kk] = load_frag(aRow + kk * 32);
#pragma unroll
  for (int kk = 0; kk < 16; ++kk) b[kk] = load_frag(bRow + kk * 32);

  v8f acc0 = {}, acc1 = {};
#pragma unroll
  for (int kk = 0; kk < 16; kk += 2) {
    acc0 = wmma16(a[kk],     b[kk],     acc0);
    acc1 = wmma16(a[kk + 1], b[kk + 1], acc1);
  }
  const v8f acc = acc0 + acc1;

  const int n = n0 + lm;
#pragma unroll
  for (int r = 0; r < 8; ++r) {
    const int    brow = m0 + r + 8 * half;                       // batch index
    const size_t xIdx = ((size_t)brow * RNN_T + t) * RNN_H + n;
    const float  v    = tanhf(acc[r] + xp[xIdx]);
    const _Float16 vh = (_Float16)v;
    hnew[(size_t)brow * RNN_H + n] = vh;
    outH[xIdx] = vh;
  }
}

__global__ void cvt_f32_f16(const float* __restrict__ s,
                            _Float16* __restrict__ d, int n)
{
  int i = blockIdx.x * blockDim.x + threadIdx.x;
  const int stride = gridDim.x * blockDim.x;
  for (; i < n; i += stride) d[i] = (_Float16)s[i];
}

__global__ void hidden_to_out(const _Float16* __restrict__ h0,
                              const _Float16* __restrict__ h1,
                              float* __restrict__ dst)
{
  const int BH = RNN_B * RNN_H;
  int i = blockIdx.x * blockDim.x + threadIdx.x;
  if (i < BH)          dst[i] = (float)h0[i];
  else if (i < 2 * BH) dst[i] = (float)h1[i - BH];
}

extern "C" void kernel_launch(void* const* d_in, const int* in_sizes, int n_in,
                              void* d_out, int out_size, void* d_ws, size_t ws_size,
                              hipStream_t stream)
{
  const float* x     = (const float*)d_in[0];
  const float* w_ih0 = (const float*)d_in[1];
  const float* w_hh0 = (const float*)d_in[2];
  const float* b_ih0 = (const float*)d_in[3];
  const float* b_hh0 = (const float*)d_in[4];
  const float* w_ih1 = (const float*)d_in[5];
  const float* w_hh1 = (const float*)d_in[6];
  const float* b_ih1 = (const float*)d_in[7];
  const float* b_hh1 = (const float*)d_in[8];
  const float* fc_w  = (const float*)d_in[9];
  const float* fc_b  = (const float*)d_in[10];
  float* out = (float*)d_out;

  // ---- workspace carving (256B aligned) ----
  char* wp = (char*)d_ws;
  auto carve = [&](size_t bytes) -> void* {
    void* r = (void*)wp;
    wp += (bytes + 255) & ~(size_t)255;
    return r;
  };
  float*    xp    = (float*)   carve((size_t)RNN_BT * RNN_H * sizeof(float));    // 64MB, reused L0/L1
  _Float16* outH  = (_Float16*)carve((size_t)RNN_BT * RNN_H * sizeof(_Float16)); // 32MB, out0 then out1
  _Float16* xh    = (_Float16*)carve((size_t)RNN_BT * RNN_I * sizeof(_Float16)); // 16MB
  _Float16* wih0h = (_Float16*)carve((size_t)RNN_H * RNN_I * sizeof(_Float16));
  _Float16* whh0h = (_Float16*)carve((size_t)RNN_H * RNN_H * sizeof(_Float16));
  _Float16* wih1h = (_Float16*)carve((size_t)RNN_H * RNN_H * sizeof(_Float16));
  _Float16* whh1h = (_Float16*)carve((size_t)RNN_H * RNN_H * sizeof(_Float16));
  _Float16* fcwh  = (_Float16*)carve((size_t)RNN_O * RNN_H * sizeof(_Float16));
  _Float16* h0a   = (_Float16*)carve((size_t)RNN_B * RNN_H * sizeof(_Float16));
  _Float16* h0b   = (_Float16*)carve((size_t)RNN_B * RNN_H * sizeof(_Float16));
  _Float16* h1a   = (_Float16*)carve((size_t)RNN_B * RNN_H * sizeof(_Float16));
  _Float16* h1b   = (_Float16*)carve((size_t)RNN_B * RNN_H * sizeof(_Float16));
  (void)ws_size; (void)in_sizes; (void)n_in; (void)out_size;

  auto cvt = [&](const float* s, _Float16* d, int n) {
    int blocks = (n + 255) / 256;
    if (blocks > 4096) blocks = 4096;
    cvt_f32_f16<<<blocks, 256, 0, stream>>>(s, d, n);
  };

  // ---- one-time f32 -> f16 conversions ----
  cvt(x,     xh,    RNN_BT * RNN_I);
  cvt(w_ih0, wih0h, RNN_H * RNN_I);
  cvt(w_hh0, whh0h, RNN_H * RNN_H);
  cvt(w_ih1, wih1h, RNN_H * RNN_H);
  cvt(w_hh1, whh1h, RNN_H * RNN_H);
  cvt(fc_w,  fcwh,  RNN_O * RNN_H);

  // ---- layer 0: input projection (one big GEMM), then sequential scan ----
  gemm_wmma_f16<<<dim3(RNN_H / 256, RNN_BT / 32), 256, 0, stream>>>(
      xh, wih0h, b_ih0, b_hh0, xp, RNN_BT, RNN_H, RNN_I);

  hipMemsetAsync(h0a, 0, (size_t)RNN_B * RNN_H * sizeof(_Float16), stream);
  _Float16* cur = h0a;
  _Float16* nxt = h0b;
  for (int t = 0; t < RNN_T; ++t) {
    rnn_step_wmma<<<dim3(RNN_H / 16), 128, 0, stream>>>(cur, whh0h, xp, t, nxt, outH);
    _Float16* tmp = cur; cur = nxt; nxt = tmp;
  }
  _Float16* h0f = cur;

  // ---- layer 1: projection of out0 (reuses xp), then scan (out1 overwrites outH) ----
  gemm_wmma_f16<<<dim3(RNN_H / 256, RNN_BT / 32), 256, 0, stream>>>(
      outH, wih1h, b_ih1, b_hh1, xp, RNN_BT, RNN_H, RNN_H);

  hipMemsetAsync(h1a, 0, (size_t)RNN_B * RNN_H * sizeof(_Float16), stream);
  cur = h1a;
  nxt = h1b;
  for (int t = 0; t < RNN_T; ++t) {
    rnn_step_wmma<<<dim3(RNN_H / 16), 128, 0, stream>>>(cur, whh1h, xp, t, nxt, outH);
    _Float16* tmp = cur; cur = nxt; nxt = tmp;
  }
  _Float16* h1f = cur;

  // ---- final FC: out = out1 @ fc_w^T + fc_b ----
  gemm_wmma_f16<<<dim3(RNN_O / 256, RNN_BT / 32), 256, 0, stream>>>(
      outH, fcwh, fc_b, nullptr, out, RNN_BT, RNN_O, RNN_H);

  // ---- hidden states -> tail of d_out ----
  hidden_to_out<<<(2 * RNN_B * RNN_H + 255) / 256, 256, 0, stream>>>(
      h0f, h1f, out + (size_t)RNN_BT * RNN_O);
}
